// TargetEncoder_31147102830959
// MI455X (gfx1250) — matrix operand order repaired and checked by reading, not compiled, not run on hardware
//
#include <hip/hip_runtime.h>

typedef __attribute__((ext_vector_type(16))) _Float16 v16h;
typedef __attribute__((ext_vector_type(8)))  float    v8f;

namespace {
constexpr int kN = 50000;
constexpr int kE = 800000;
}

// ---------------- graph preprocessing ----------------

__global__ void cheb_deg_kernel(const int* __restrict__ src, const int* __restrict__ tgt,
                                float* __restrict__ deg, int E) {
    int e = blockIdx.x * blockDim.x + threadIdx.x;
    if (e < E) {
        int s = src[e];
        if (s != tgt[e]) atomicAdd(&deg[s], 1.0f);
    }
}

__global__ void cheb_dis_kernel(float* __restrict__ deg, int n) {
    int i = blockIdx.x * blockDim.x + threadIdx.x;
    if (i < n) {
        float d = deg[i];
        deg[i] = (d > 0.0f) ? rsqrtf(fmaxf(d, 1.0f)) : 0.0f;
    }
}

__global__ void cheb_edgew_kernel(const int* __restrict__ src, const int* __restrict__ tgt,
                                  const float* __restrict__ dis, float* __restrict__ w, int E) {
    int e = blockIdx.x * blockDim.x + threadIdx.x;
    if (e < E) {
        int s = src[e], t = tgt[e];
        w[e] = (s != t) ? (-dis[s] * dis[t]) : 0.0f;
    }
}

// ---------------- sparse propagation: out[tgt] += w * t[src] ----------------
// F/4 consecutive threads per edge; b128 gathers, coalesced f32 atomics.

__global__ void cheb_prop_kernel(const float* __restrict__ t, float* __restrict__ out,
                                 const int* __restrict__ src, const int* __restrict__ tgt,
                                 const float* __restrict__ w, int E, int logF4) {
    int idx = blockIdx.x * blockDim.x + threadIdx.x;
    if (idx >= (E << logF4)) return;
    int e  = idx >> logF4;
    int f4 = (idx & ((1 << logF4) - 1)) << 2;
    const int logF = logF4 + 2;
    float we = w[e];
    if (we != 0.0f) {
        float4 v = *(const float4*)(t + ((size_t)src[e] << logF) + f4);
        float* o = out + ((size_t)tgt[e] << logF) + f4;
        atomicAdd(o + 0, we * v.x);
        atomicAdd(o + 1, we * v.y);
        atomicAdd(o + 2, we * v.z);
        atomicAdd(o + 3, we * v.w);
    }
}

// tx2 = 2*tx2 - tx0  (Chebyshev recursion closure)
__global__ void cheb_combine_kernel(float* __restrict__ tx2, const float* __restrict__ tx0, int n) {
    int i = blockIdx.x * blockDim.x + threadIdx.x;
    if (i < n) tx2[i] = 2.0f * tx2[i] - tx0[i];
}

// ---------------- W pre-pack: f32 [3][Fin][Fout] -> f16 B-fragment layout ----------------
// Layout: [(t*ktiles + kt)*ntiles + nt][lane 0..31][i 0..15], where per lane:
//   col = nt*16 + (lane&15), K = kt*32 + (i>>3)*16 + (lane>>4)*8 + (i&7)
// so a wave's B operand for one 32x16 tile is 32 contiguous bytes per lane.

__global__ void cheb_packW_kernel(const float* __restrict__ W, _Float16* __restrict__ Wh,
                                  int Fin, int Fout) {
    int idx = blockIdx.x * blockDim.x + threadIdx.x;
    if (idx >= 3 * Fin * Fout) return;
    const int ntiles = Fout >> 4;
    const int ktiles = Fin >> 5;
    int i    = idx & 15;
    int lane = (idx >> 4) & 31;
    int tile = idx >> 9;
    int nt  = tile % ntiles;
    int ktt = tile / ntiles;
    int kt  = ktt % ktiles;
    int t   = ktt / ktiles;
    int k = kt * 32 + (i >> 3) * 16 + (lane >> 4) * 8 + (i & 7);
    int n = nt * 16 + (lane & 15);
    Wh[idx] = (_Float16)W[((size_t)t * Fin + k) * Fout + n];
}

// ---------------- dense: out = relu(t0@W0 + t1@W1 + t2@W2 + b) via WMMA ----------------
// One wave per 16(M)x16(N) output tile; 8 waves/block span M so every wave's A tile is
// unique (no duplicated A traffic) while B fragments come from the tiny packed buffer.
// A (16x32 f16): lane = row (lane&15); lane halves split K {0..7,16..23} vs {8..15,24..31}.
// B: single contiguous v16h per lane from the packed layout (2x global_load_b128).
// C/D (16x16 f32): acc[j] -> row = j + 8*(lane>=16), col = lane&15.

__global__ __launch_bounds__(256)
void cheb_gemm_relu_wmma(const float* __restrict__ t0, const float* __restrict__ t1,
                         const float* __restrict__ t2, const _Float16* __restrict__ Wh,
                         const float* __restrict__ bias, float* __restrict__ out,
                         int Fin, int Fout, int nMtiles) {
    const int lane = threadIdx.x & 31;
    const int warp = threadIdx.x >> 5;
    const int mt   = blockIdx.x * 8 + warp;
    if (mt >= nMtiles) return;                // wave-uniform exit: EXEC stays all-ones

    const int nt    = blockIdx.y;
    const int r     = lane & 15;
    const int hi    = lane >> 4;
    const int m0    = mt * 16;
    const int n0    = nt * 16;
    const int m     = m0 + r;
    const int col   = n0 + r;
    const int koffA = hi * 8;
    const int ktiles = Fin >> 5;
    const int ntiles = Fout >> 4;

    v8f acc = {};

#pragma unroll
    for (int tm = 0; tm < 3; ++tm) {
        const float* __restrict__ T =
            (tm == 0) ? t0 : ((tm == 1) ? t1 : t2);
        const _Float16* __restrict__ Wt = Wh + (size_t)tm * Fin * Fout;
        for (int kt = 0; kt < ktiles; ++kt) {
            // ---- A tile: 4x b128 loads per lane, downcast f32 -> f16 ----
            const float* arow = T + (size_t)m * Fin + kt * 32 + koffA;
            float4 a0 = *(const float4*)(arow);        // K = koffA + 0..3
            float4 a1 = *(const float4*)(arow + 4);    // K = koffA + 4..7
            float4 a2 = *(const float4*)(arow + 16);   // K = 16 + koffA + 0..3
            float4 a3 = *(const float4*)(arow + 20);   // K = 16 + koffA + 4..7
            v16h a;
            a[0]  = (_Float16)a0.x; a[1]  = (_Float16)a0.y; a[2]  = (_Float16)a0.z; a[3]  = (_Float16)a0.w;
            a[4]  = (_Float16)a1.x; a[5]  = (_Float16)a1.y; a[6]  = (_Float16)a1.z; a[7]  = (_Float16)a1.w;
            a[8]  = (_Float16)a2.x; a[9]  = (_Float16)a2.y; a[10] = (_Float16)a2.z; a[11] = (_Float16)a2.w;
            a[12] = (_Float16)a3.x; a[13] = (_Float16)a3.y; a[14] = (_Float16)a3.z; a[15] = (_Float16)a3.w;

            // ---- B tile: one contiguous 32B fragment per lane ----
            v16h b = *(const v16h*)(Wt + (((size_t)(kt * ntiles + nt) * 32) + lane) * 16);

            acc = __builtin_amdgcn_wmma_f32_16x16x32_f16(
                /*neg_a=*/false, a, /*neg_b=*/false, b,
                /*c_mod=*/(short)0, acc, /*reuse_a=*/false, /*reuse_b=*/false);
        }
    }

    const float bb = bias[col];
    const int rbase = m0 + hi * 8;
#pragma unroll
    for (int j = 0; j < 8; ++j) {
        float v = acc[j] + bb;
        out[(size_t)(rbase + j) * Fout + col] = (v > 0.0f) ? v : 0.0f;
    }
}

// ---------------- driver ----------------

extern "C" void kernel_launch(void* const* d_in, const int* in_sizes, int n_in,
                              void* d_out, int out_size, void* d_ws, size_t ws_size,
                              hipStream_t stream) {
    (void)in_sizes; (void)n_in; (void)out_size; (void)ws_size;

    const float* x  = (const float*)d_in[0];
    const int*   ei = (const int*)  d_in[1];
    const float* W1 = (const float*)d_in[2];
    const float* b1 = (const float*)d_in[3];
    const float* W2 = (const float*)d_in[4];
    const float* b2 = (const float*)d_in[5];
    const float* W3 = (const float*)d_in[6];
    const float* b3 = (const float*)d_in[7];
    const int* src = ei;
    const int* tgt = ei + kE;

    float* wE  = (float*)d_ws;                       // E
    float* dis = wE  + kE;                           // N (deg -> dis in place)
    float* tx1 = dis + kN;                           // N*256
    float* tx2 = tx1 + (size_t)kN * 256;             // N*256
    float* h1  = tx2 + (size_t)kN * 256;             // N*128
    float* h2  = h1  + (size_t)kN * 128;             // N*256
    _Float16* Wh = (_Float16*)(h2 + (size_t)kN * 256); // 3*256*256 halves max (32B-aligned)
    float* out = (float*)d_out;

    // scaled-Laplacian edge weights
    hipMemsetAsync(dis, 0, kN * sizeof(float), stream);
    cheb_deg_kernel  <<<(kE + 255) / 256, 256, 0, stream>>>(src, tgt, dis, kE);
    cheb_dis_kernel  <<<(kN + 255) / 256, 256, 0, stream>>>(dis, kN);
    cheb_edgew_kernel<<<(kE + 255) / 256, 256, 0, stream>>>(src, tgt, dis, wE, kE);

    struct Layer { const float* X; int Fin; int logF4; const float* W; const float* b; float* O; int Fout; };
    Layer layers[3] = {
        { x,   64, 4, W1, b1, h1, 128 },
        { h1, 128, 5, W2, b2, h2, 256 },
        { h2, 256, 6, W3, b3, out, 256 },
    };

    const int nMtiles = kN / 16;                     // 3125
    const int gemmBlocksX = (nMtiles + 7) / 8;       // 8 M-tiles (waves) per block

    for (int l = 0; l < 3; ++l) {
        const Layer& L = layers[l];
        const size_t nf = (size_t)kN * L.Fin;
        const int totalEF4 = kE << L.logF4;

        hipMemsetAsync(tx1, 0, nf * sizeof(float), stream);
        cheb_prop_kernel<<<(totalEF4 + 255) / 256, 256, 0, stream>>>(L.X, tx1, src, tgt, wE, kE, L.logF4);

        hipMemsetAsync(tx2, 0, nf * sizeof(float), stream);
        cheb_prop_kernel<<<(totalEF4 + 255) / 256, 256, 0, stream>>>(tx1, tx2, src, tgt, wE, kE, L.logF4);
        cheb_combine_kernel<<<((int)nf + 255) / 256, 256, 0, stream>>>(tx2, L.X, (int)nf);

        const int wTotal = 3 * L.Fin * L.Fout;
        cheb_packW_kernel<<<(wTotal + 255) / 256, 256, 0, stream>>>(L.W, Wh, L.Fin, L.Fout);

        dim3 grid(gemmBlocksX, L.Fout / 16);
        cheb_gemm_relu_wmma<<<grid, 256, 0, stream>>>(L.X, tx1, tx2, Wh, L.b, L.O,
                                                      L.Fin, L.Fout, nMtiles);
    }
}
